// MHANet_10754598109572
// MI455X (gfx1250) — matrix operand order repaired and checked by reading, not compiled, not run on hardware
//
#include <hip/hip_runtime.h>
#include <math.h>

typedef _Float16 f16;
typedef __attribute__((ext_vector_type(16))) _Float16 v16h;
typedef __attribute__((ext_vector_type(8)))  float    v8f;

#define BNS 0.9999950000374997f  // 1/sqrt(1+1e-5)

// ---------------- sizes ----------------
#define B_   16
#define T_   1024
#define HID_ 128
#define NH_  8
#define HD_  16
#define CIN_ 64

// ---------------- ws layout (float units) ----------------
#define OFF_SHIFT   ((size_t)0)
#define OFF_SCALE   ((size_t)2048)
#define OFF_GATE    ((size_t)4096)
#define OFF_SH4     ((size_t)6144)
#define OFF_SC24    ((size_t)6208)
#define OFF_FEAT    ((size_t)6272)
#define OFF_AVAL    ((size_t)6336)
#define OFF_W16Q    ((size_t)6400)       // 640*128 halves   = 40960 float-units
#define OFF_W16O    ((size_t)47360)      // 128*384 halves   = 24576
#define OFF_XH      ((size_t)71936)      // 2097152
#define OFF_XMOD16  ((size_t)2169088)    // 16*1024*128 halves = 1048576
#define OFF_VRAW    ((size_t)3217664)    // 2097152
#define OFF_VP      ((size_t)5314816)    // 49152
#define OFF_Q16     ((size_t)5363968)    // 1048576
#define OFF_K16     ((size_t)6412544)    // 1048576
#define OFF_V16     ((size_t)7461120)    // 1048576
#define OFF_U16     ((size_t)8509696)    // 16*1024*384 halves = 3145728
#define OFF_OUTB    ((size_t)11655424)   // 2097152
#define OFF_XBLK    ((size_t)13752576)   // 1048576
#define OFF_SP      ((size_t)14801152)   // 49152
#define OFF_XMGA    ((size_t)14850304)   // 1048576 -> total 15898880 floats (~63.6 MB)
#define OFF_TBUF    OFF_XMOD16           // aliases xmod16/vraw/vp/q16 (dead by k_tconv); 4198400 floats

__device__ __forceinline__ float siluf(float x){ return x / (1.f + __expf(-x)); }
__device__ __forceinline__ float eluf(float x){ return x > 0.f ? x : expm1f(x); }
__device__ __forceinline__ float geluf(float x){ return 0.5f * x * (1.f + erff(x * 0.70710678118654752f)); }

// ================= K1: cond path =================
__global__ __launch_bounds__(128) void k_cond(
    const float* __restrict__ cond, const float* __restrict__ ce_w1, const float* __restrict__ ce_b1,
    const float* __restrict__ ce_w2, const float* __restrict__ ce_b2,
    const float* __restrict__ mod_w, const float* __restrict__ mod_b,
    const float* __restrict__ fcp_w, const float* __restrict__ fcp_b,
    const float* __restrict__ ada_w, const float* __restrict__ ada_b,
    float* __restrict__ shift, float* __restrict__ scale, float* __restrict__ gate,
    float* __restrict__ sh4, float* __restrict__ sc24) {
  __shared__ float h1[128], mbuf[128];
  const int tid = threadIdx.x;
  for (int b = 0; b < B_; ++b) {
    const float* cv = cond + b * 32;
    float a = ce_b1[tid];
    for (int i = 0; i < 32; ++i) a += ce_w1[tid * 32 + i] * cv[i];
    h1[tid] = siluf(a);
    __syncthreads();
    float c2 = ce_b2[tid];
    for (int i = 0; i < 128; ++i) c2 += ce_w2[tid * 128 + i] * h1[i];
    mbuf[tid] = siluf(c2);
    __syncthreads();
    for (int r5 = 0; r5 < 3; ++r5) {
      const int r = tid + r5 * 128;
      float v = mod_b[r];
      const float* wr = mod_w + (size_t)r * 128;
      for (int i = 0; i < 128; ++i) v += wr[i] * mbuf[i];
      if (r5 == 0)      shift[b * 128 + tid] = v;
      else if (r5 == 1) scale[b * 128 + tid] = v;
      else              gate[b * 128 + tid]  = 1.f / (1.f + __expf(-v));
    }
    if (tid < 8) {
      float c4[4];
      for (int i = 0; i < 4; ++i) {
        float s = fcp_b[i];
        for (int k = 0; k < 32; ++k) s += fcp_w[i * 32 + k] * cv[k];
        c4[i] = siluf(s);
      }
      float v = ada_b[tid];
      for (int i = 0; i < 4; ++i) v += ada_w[tid * 4 + i] * c4[i];
      if (tid < 4) sh4[b * 4 + tid] = v;
      else         sc24[b * 4 + tid - 4] = v;
    }
    __syncthreads();
  }
}

// ================= K1b: convert GEMM weights to f16 =================
__global__ __launch_bounds__(256) void k_w16(
    const float* __restrict__ qkv_w, const float* __restrict__ op_w,
    f16* __restrict__ wq16, f16* __restrict__ wo16) {
  const int gid = blockIdx.x * blockDim.x + threadIdx.x;
  if (gid < 640 * 128)             wq16[gid] = (f16)qkv_w[gid];
  else if (gid < 640 * 128 + 128 * 384) {
    const int g2 = gid - 640 * 128;
    wo16[g2] = (f16)op_w[g2];
  }
}

// ======== K2: column kernel: cp GEMV -> x_h, channel LN + modulation -> xmod16 ========
__global__ __launch_bounds__(128) void k_ln(
    const float* __restrict__ x, const float* __restrict__ cp_w, const float* __restrict__ cp_b,
    const float* __restrict__ ln_g, const float* __restrict__ ln_b,
    const float* __restrict__ shift, const float* __restrict__ scale,
    float* __restrict__ xh, f16* __restrict__ xmod16) {
  __shared__ float xcol[64], red[128];
  const int t = blockIdx.x & (T_ - 1), b = blockIdx.x >> 10, tid = threadIdx.x;
  if (tid < 64) xcol[tid] = x[((size_t)(b * 64 + tid)) * T_ + t];
  __syncthreads();
  float h = cp_b[tid];
  {
    const float* wr = cp_w + (size_t)tid * 64;
    for (int i = 0; i < 64; ++i) h += wr[i] * xcol[i];
  }
  xh[((size_t)(b * 128 + tid)) * T_ + t] = h;
  red[tid] = h; __syncthreads();
  for (int s = 64; s > 0; s >>= 1) { if (tid < s) red[tid] += red[tid + s]; __syncthreads(); }
  const float mean = red[0] * (1.f / 128.f); __syncthreads();
  const float d0 = h - mean;
  red[tid] = d0 * d0; __syncthreads();
  for (int s = 64; s > 0; s >>= 1) { if (tid < s) red[tid] += red[tid + s]; __syncthreads(); }
  const float rstd = rsqrtf(red[0] * (1.f / 128.f) + 1e-5f);
  const float xm = (d0 * rstd * ln_g[tid] + ln_b[tid]) * (1.f + scale[b * 128 + tid]) + shift[b * 128 + tid];
  xmod16[((size_t)(b * 1024) + t) * 128 + tid] = (f16)xm;
}

// ======== K3: qkv GEMM via WMMA; fused head-LN (q/k), vraw (v), gelu->u16 (mlp) ========
__global__ __launch_bounds__(256) void k_qkvmm(
    const f16* __restrict__ wq16, const f16* __restrict__ xmod16, const float* __restrict__ qkv_b,
    const float* __restrict__ qn_g, const float* __restrict__ qn_b,
    const float* __restrict__ kn_g, const float* __restrict__ kn_b,
    float* __restrict__ vraw, f16* __restrict__ q16, f16* __restrict__ k16, f16* __restrict__ u16) {
  const int lane = threadIdx.x & 31, wave = threadIdx.x >> 5;
  const int ln16 = lane & 15, hiA = lane >> 4;
  int id = blockIdx.x * 8 + wave;
  const int mt = id % 40; id /= 40;
  const int tt = id & 63; const int b = id >> 6;   // 16*64*40 waves
  const int m0 = mt * 16, t0 = tt * 16;

  v8f acc = {0.f, 0.f, 0.f, 0.f, 0.f, 0.f, 0.f, 0.f};
  const f16* wrow  = wq16 + (size_t)(m0 + ln16) * 128;
  const f16* xbase = xmod16 + ((size_t)(b * 1024) + t0 + ln16) * 128;
  #pragma unroll
  for (int kc = 0; kc < 128; kc += 32) {
    v16h aW, bX;
    #pragma unroll
    for (int j = 0; j < 16; ++j) aW[j] = wrow[kc + 16 * hiA + j];
    #pragma unroll
    for (int j = 0; j < 16; ++j) bX[j] = xbase[kc + 16 * hiA + j];
    acc = __builtin_amdgcn_wmma_f32_16x16x32_f16(false, aW, false, bX, (short)0, acc, false, false);
  }
  #pragma unroll
  for (int g = 0; g < 8; ++g) acc[g] += qkv_b[m0 + g + 8 * hiA];

  const int tcol = t0 + ln16;
  if (m0 < 256) {
    // q/k: per-head LayerNorm over d=16 (one tile == one head), in-register
    float s = 0.f, s2 = 0.f;
    #pragma unroll
    for (int g = 0; g < 8; ++g) { s += acc[g]; s2 += acc[g] * acc[g]; }
    s  += __shfl_xor(s, 16, 32);
    s2 += __shfl_xor(s2, 16, 32);
    const float mean = s * (1.f / 16.f);
    const float rstd = rsqrtf(s2 * (1.f / 16.f) - mean * mean + 1e-5f);
    const int isK = (m0 >= 128);
    const int hh = (m0 & 127) >> 4;
    const float* gg = isK ? kn_g : qn_g;
    const float* bv = isK ? kn_b : qn_b;
    f16* dst = (isK ? k16 : q16) + (((size_t)(b * 8 + hh)) * T_ + tcol) * 16;
    #pragma unroll
    for (int g = 0; g < 8; ++g) {
      const int d = g + 8 * hiA;
      dst[d] = (f16)((acc[g] - mean) * rstd * gg[d] + bv[d]);
    }
  } else if (m0 < 384) {
    #pragma unroll
    for (int g = 0; g < 8; ++g)
      vraw[((size_t)(b * 128 + (m0 - 256) + g + 8 * hiA)) * T_ + tcol] = acc[g];
  } else {
    f16* dst = u16 + ((size_t)(b * 1024) + tcol) * 384 + 128 + (m0 - 384);
    #pragma unroll
    for (int g = 0; g < 8; ++g) dst[g + 8 * hiA] = (f16)geluf(acc[g]);
  }
}

// ================= K4a: vp = mta_in @ vraw =================
__global__ __launch_bounds__(256) void k_vp(
    const float* __restrict__ vraw, const float* __restrict__ mta_in_w,
    const float* __restrict__ mta_in_b, float* __restrict__ vp) {
  const int gid = blockIdx.x * blockDim.x + threadIdx.x;
  if (gid >= B_ * 3 * T_) return;
  const int t = gid & (T_ - 1), j = (gid >> 10) % 3, b = gid / (3 * T_);
  float a = mta_in_b[j];
  const float* col = vraw + (size_t)b * 128 * T_ + t;
  const float* w = mta_in_w + j * 128;
  for (int i = 0; i < 128; ++i) a += w[i] * col[(size_t)i * T_];
  vp[(size_t)gid] = a;
}

// ================= K4b: mt conv1d + time-LN + elu + mean =================
__global__ __launch_bounds__(256) void k_ascalar(
    const float* __restrict__ vp, const float* __restrict__ w2, const float* __restrict__ b2,
    const float* __restrict__ w4, const float* __restrict__ b4,
    const float* __restrict__ w6, const float* __restrict__ b6, float* __restrict__ avals) {
  __shared__ float red[256];
  const int j = blockIdx.x % 3, b = blockIdx.x / 3, tid = threadIdx.x;
  const float* w = (j == 0) ? w2 : ((j == 1) ? w4 : w6);
  const float bias = (j == 0) ? b2[0] : ((j == 1) ? b4[0] : b6[0]);
  const int ksz = 2 * j + 2, pad = j;
  const float* src = vp + (size_t)(b * 3 + j) * T_;
  float s1 = 0.f, s2 = 0.f;
  for (int tp = tid; tp < 1023; tp += 256) {
    float y = bias;
    for (int k = 0; k < ksz; ++k) {
      const int u = tp - pad + k;
      y += w[k] * ((u >= 0 && u < T_) ? src[u] : 0.f);
    }
    s1 += y; s2 += y * y;
  }
  red[tid] = s1; __syncthreads();
  for (int s = 128; s > 0; s >>= 1) { if (tid < s) red[tid] += red[tid + s]; __syncthreads(); }
  const float mean = red[0] / 1023.f; __syncthreads();
  red[tid] = s2; __syncthreads();
  for (int s = 128; s > 0; s >>= 1) { if (tid < s) red[tid] += red[tid + s]; __syncthreads(); }
  const float rstd = rsqrtf(red[0] / 1023.f - mean * mean + 1e-5f); __syncthreads();
  float se = 0.f;
  for (int tp = tid; tp < 1023; tp += 256) {
    float y = bias;
    for (int k = 0; k < ksz; ++k) {
      const int u = tp - pad + k;
      y += w[k] * ((u >= 0 && u < T_) ? src[u] : 0.f);
    }
    se += eluf((y - mean) * rstd);
  }
  red[tid] = se; __syncthreads();
  for (int s = 128; s > 0; s >>= 1) { if (tid < s) red[tid] += red[tid + s]; __syncthreads(); }
  if (tid == 0) avals[b * 3 + j] = red[0] / 1023.f;
}

// ================= K4c: v update -> f16 =================
__global__ __launch_bounds__(256) void k_vupd(
    const float* __restrict__ vraw, const float* __restrict__ vp, const float* __restrict__ avals,
    const float* __restrict__ mta_out_w, const float* __restrict__ mta_out_b, f16* __restrict__ v16) {
  const int gid = blockIdx.x * blockDim.x + threadIdx.x;
  if (gid >= B_ * HID_ * T_) return;
  const int t = gid & (T_ - 1), c = (gid >> 10) & 127, b = gid >> 17;
  const float g0 = avals[b * 3 + 0] * vp[((size_t)(b * 3 + 0)) * T_ + t];
  const float g1 = avals[b * 3 + 1] * vp[((size_t)(b * 3 + 1)) * T_ + t];
  const float g2 = avals[b * 3 + 2] * vp[((size_t)(b * 3 + 2)) * T_ + t];
  const float vn = vraw[(size_t)gid] + mta_out_b[c]
                 + mta_out_w[c * 3 + 0] * g0 + mta_out_w[c * 3 + 1] * g1 + mta_out_w[c * 3 + 2] * g2;
  v16[(((size_t)(b * 8 + (c >> 4))) * T_ + t) * 16 + (c & 15)] = (f16)vn;
}

// ================= K5: flash attention via WMMA f16 (wave32), writes f16 into u16[k<128] =================
__global__ __launch_bounds__(256) void k_attn(
    const f16* __restrict__ q16, const f16* __restrict__ k16,
    const f16* __restrict__ v16, f16* __restrict__ u16) {
  __shared__ f16 lds_p[8][16][32];
  const int lane = threadIdx.x & 31;
  const int wave = threadIdx.x >> 5;
  const int ln16 = lane & 15;
  const int hiA  = lane >> 4;
  int idx = blockIdx.x;
  const int qt8 = idx & 7; idx >>= 3;
  const int h = idx & 7;   idx >>= 3;
  const int b = idx;
  const int q0 = (qt8 * 8 + wave) * 16;
  const size_t base = ((size_t)(b * 8 + h)) * T_ * 16;

  v16h aQ;
  {
    const f16* qp = q16 + base + (size_t)(q0 + ln16) * 16;
    #pragma unroll
    for (int j = 0; j < 16; ++j) aQ[j] = hiA ? (f16)0.f : qp[j];
  }
  v8f acc = {0.f, 0.f, 0.f, 0.f, 0.f, 0.f, 0.f, 0.f};
  float mrow[8], lrow[8];
  #pragma unroll
  for (int g = 0; g < 8; ++g) { mrow[g] = -3.0e38f; lrow[g] = 0.f; }

  for (int kc = 0; kc < T_ / 32; ++kc) {
    const int k0 = kc * 32;
    v16h bK0, bK1;
    if (hiA == 0) {
      const f16* kp0 = k16 + base + (size_t)(k0 + ln16) * 16;
      const f16* kp1 = k16 + base + (size_t)(k0 + 16 + ln16) * 16;
      #pragma unroll
      for (int j = 0; j < 16; ++j) { bK0[j] = kp0[j]; bK1[j] = kp1[j]; }
    } else {
      #pragma unroll
      for (int j = 0; j < 16; ++j) { bK0[j] = (f16)0.f; bK1[j] = (f16)0.f; }
    }
    const v8f z = {0.f, 0.f, 0.f, 0.f, 0.f, 0.f, 0.f, 0.f};
    v8f s0 = __builtin_amdgcn_wmma_f32_16x16x32_f16(false, aQ, false, bK0, (short)0, z, false, false);
    v8f s1 = __builtin_amdgcn_wmma_f32_16x16x32_f16(false, aQ, false, bK1, (short)0, z, false, false);

    #pragma unroll
    for (int g = 0; g < 8; ++g) {
      const float v0 = s0[g] * 0.25f;
      const float v1 = s1[g] * 0.25f;
      float cm = fmaxf(v0, v1);
      cm = fmaxf(cm, __shfl_xor(cm, 1, 32));
      cm = fmaxf(cm, __shfl_xor(cm, 2, 32));
      cm = fmaxf(cm, __shfl_xor(cm, 4, 32));
      cm = fmaxf(cm, __shfl_xor(cm, 8, 32));
      const float mnew = fmaxf(mrow[g], cm);
      const float alpha = __expf(mrow[g] - mnew);
      const float p0 = __expf(v0 - mnew);
      const float p1 = __expf(v1 - mnew);
      float rs = p0 + p1;
      rs += __shfl_xor(rs, 1, 32);
      rs += __shfl_xor(rs, 2, 32);
      rs += __shfl_xor(rs, 4, 32);
      rs += __shfl_xor(rs, 8, 32);
      lrow[g] = lrow[g] * alpha + rs;
      mrow[g] = mnew;
      acc[g] *= alpha;
      lds_p[wave][g + 8 * hiA][ln16]      = (f16)p0;
      lds_p[wave][g + 8 * hiA][16 + ln16] = (f16)p1;
    }
    __asm__ volatile("s_wait_dscnt 0x0" ::: "memory");
    v16h aP;
    #pragma unroll
    for (int j = 0; j < 16; ++j) {
      const int key = j + hiA * 16;
      aP[j] = lds_p[wave][ln16][key];
    }
    v16h bV;
    {
      const f16* vpx = v16 + base + (size_t)(k0 + 16 * hiA) * 16 + ln16;
      #pragma unroll
      for (int j = 0; j < 16; ++j) bV[j] = vpx[(size_t)j * 16];
    }
    acc = __builtin_amdgcn_wmma_f32_16x16x32_f16(false, aP, false, bV, (short)0, acc, false, false);
  }
  #pragma unroll
  for (int g = 0; g < 8; ++g) {
    u16[((size_t)(b * 1024) + (q0 + g + 8 * hiA)) * 384 + h * 16 + ln16] = (f16)(acc[g] / lrow[g]);
  }
}

// ================= K6a: op conv (128x384) via WMMA =================
__global__ __launch_bounds__(256) void k_opmm(
    const f16* __restrict__ wo16, const f16* __restrict__ u16, const float* __restrict__ op_b,
    float* __restrict__ outb) {
  const int lane = threadIdx.x & 31, wave = threadIdx.x >> 5;
  const int ln16 = lane & 15, hiA = lane >> 4;
  int id = blockIdx.x * 8 + wave;
  const int mt = id & 7; id >>= 3;
  const int tt = id & 63; const int b = id >> 6;   // 16*64*8 waves
  const int m0 = mt * 16, t0 = tt * 16;

  v8f acc = {0.f, 0.f, 0.f, 0.f, 0.f, 0.f, 0.f, 0.f};
  const f16* wrow  = wo16 + (size_t)(m0 + ln16) * 384;
  const f16* ubase = u16 + ((size_t)(b * 1024) + t0 + ln16) * 384;
  #pragma unroll
  for (int kc = 0; kc < 384; kc += 32) {
    v16h aW, bU;
    #pragma unroll
    for (int j = 0; j < 16; ++j) aW[j] = wrow[kc + 16 * hiA + j];
    #pragma unroll
    for (int j = 0; j < 16; ++j) bU[j] = ubase[kc + 16 * hiA + j];
    acc = __builtin_amdgcn_wmma_f32_16x16x32_f16(false, aW, false, bU, (short)0, acc, false, false);
  }
  const int tcol = t0 + ln16;
  #pragma unroll
  for (int g = 0; g < 8; ++g)
    outb[((size_t)(b * 128 + m0 + g + 8 * hiA)) * T_ + tcol] = acc[g] + op_b[m0 + g + 8 * hiA];
}

// ================= K6b: gate + fin conv =================
__global__ __launch_bounds__(128) void k_out2(
    const float* __restrict__ outb, const float* __restrict__ xh, const float* __restrict__ gateact,
    const float* __restrict__ fin_w, const float* __restrict__ fin_b, float* __restrict__ xblk) {
  __shared__ float xin[128];
  const int t = blockIdx.x & (T_ - 1), b = blockIdx.x >> 10, tid = threadIdx.x;
  xin[tid] = xh[((size_t)(b * 128 + tid)) * T_ + t]
           + gateact[b * 128 + tid] * outb[((size_t)(b * 128 + tid)) * T_ + t];
  __syncthreads();
  if (tid < 64) {
    float a = fin_b[tid];
    const float* wr = fin_w + (size_t)tid * 128;
    for (int i = 0; i < 128; ++i) a += wr[i] * xin[i];
    xblk[((size_t)(b * 64 + tid)) * T_ + t] = a;
  }
}

// ================= K7a: sp = bn(dc conv) =================
__global__ __launch_bounds__(256) void k_sp(
    const float* __restrict__ xblk, const float* __restrict__ dc_w, const float* __restrict__ dc_b,
    const float* __restrict__ g, const float* __restrict__ bb, float* __restrict__ sp) {
  const int gid = blockIdx.x * blockDim.x + threadIdx.x;
  if (gid >= B_ * 3 * T_) return;
  const int t = gid & (T_ - 1), j = (gid >> 10) % 3, b = gid / (3 * T_);
  float a = dc_b[j];
  const float* col = xblk + (size_t)b * 64 * T_ + t;
  const float* w = dc_w + j * 64;
  for (int i = 0; i < 64; ++i) a += w[i] * col[(size_t)i * T_];
  sp[(size_t)gid] = a * (g[j] * BNS) + bb[j];
}

// ================= K7b: gated dilated mixes + uc + residual =================
__global__ __launch_bounds__(256) void k_mga(
    const float* __restrict__ sp, const float* __restrict__ c0_w, const float* __restrict__ c0_b,
    const float* __restrict__ c1_w, const float* __restrict__ c1_b,
    const float* __restrict__ c2_w, const float* __restrict__ c2_b,
    const float* __restrict__ uc_w, const float* __restrict__ uc_b,
    const float* __restrict__ xblk, float* __restrict__ xmga) {
  const int gid = blockIdx.x * blockDim.x + threadIdx.x;
  if (gid >= B_ * T_) return;
  const int t = gid & (T_ - 1), b = gid >> 10;
  const float* s0 = sp + (size_t)(b * 3 + 0) * T_;
  const float* s1 = sp + (size_t)(b * 3 + 1) * T_;
  const float* s2 = sp + (size_t)(b * 3 + 2) * T_;
  float a0 = c0_b[0];
  for (int i = 0; i < 3; ++i) { const int u = t - 1 + i; a0 += c0_w[3 + i] * ((u >= 0 && u < T_) ? s0[u] : 0.f); }
  a0 *= s0[t];
  float a1 = c1_b[0];
  for (int i = 0; i < 5; ++i) { const int u = t - 4 + 2 * i; a1 += c1_w[10 + i] * ((u >= 0 && u < T_) ? s1[u] : 0.f); }
  a1 *= s1[t];
  float a2 = c2_b[0];
  for (int i = 0; i < 7; ++i) { const int u = t - 9 + 3 * i; a2 += c2_w[21 + i] * ((u >= 0 && u < T_) ? s2[u] : 0.f); }
  a2 *= s2[t];
  for (int c = 0; c < 64; ++c) {
    const size_t o = ((size_t)(b * 64 + c)) * T_ + t;
    xmga[o] = uc_b[c] + uc_w[c * 3] * a0 + uc_w[c * 3 + 1] * a1 + uc_w[c * 3 + 2] * a2 + xblk[o];
  }
}

// ================= K8: t-conv (1x8, pad 4 in W) + bn + elu =================
__global__ __launch_bounds__(256) void k_tconv(
    const float* __restrict__ xmga, const float* __restrict__ tc_w, const float* __restrict__ tc_b,
    const float* __restrict__ bg, const float* __restrict__ bb, float* __restrict__ tbuf) {
  const int gid = blockIdx.x * blockDim.x + threadIdx.x;
  if (gid >= B_ * 64 * 1025) return;
  const int w = gid % 1025, hi = (gid / 1025) & 63, b = gid / (1025 * 64);
  float win[8];
  const float* row = xmga + (size_t)(b * 64 + hi) * T_;
  #pragma unroll
  for (int k = 0; k < 8; ++k) { const int u = w - 4 + k; win[k] = (u >= 0 && u < T_) ? row[u] : 0.f; }
  #pragma unroll
  for (int oc = 0; oc < 4; ++oc) {
    float a = tc_b[oc];
    #pragma unroll
    for (int k = 0; k < 8; ++k) a += tc_w[oc * 8 + k] * win[k];
    tbuf[(((size_t)(b * 4 + oc)) * 64 + hi) * 1025 + w] = eluf(a * (bg[oc] * BNS) + bb[oc]);
  }
}

// ================= K9: s-conv (64x1) + bn + elu + mean -> feat (deterministic) =================
__global__ __launch_bounds__(256) void k_sconv(
    const float* __restrict__ tbuf, const float* __restrict__ sc_w, const float* __restrict__ sc_b,
    const float* __restrict__ bg, const float* __restrict__ bb, float* __restrict__ feat) {
  __shared__ float red[256];
  const int oc = blockIdx.x & 3, b = blockIdx.x >> 2, tid = threadIdx.x;
  float acc = 0.f;
  for (int w = tid; w < 1025; w += 256) {
    float a = sc_b[oc];
    for (int ic = 0; ic < 4; ++ic) {
      const float* tb = tbuf + (((size_t)(b * 4 + ic)) * 64) * 1025 + w;
      const float* ww = sc_w + (oc * 4 + ic) * 64;
      for (int kh = 0; kh < 64; ++kh) a += ww[kh] * tb[(size_t)kh * 1025];
    }
    acc += eluf(a * (bg[oc] * BNS) + bb[oc]);
  }
  red[tid] = acc; __syncthreads();
  for (int s = 128; s > 0; s >>= 1) { if (tid < s) red[tid] += red[tid + s]; __syncthreads(); }
  if (tid == 0) feat[b * 4 + oc] = red[0] * (1.f / 1025.f);
}

// ================= K10: final LN + ada + linear =================
__global__ void k_final(
    const float* __restrict__ feat, const float* __restrict__ sh4, const float* __restrict__ sc24,
    const float* __restrict__ flin_w, const float* __restrict__ flin_b, float* __restrict__ out) {
  const int tid = threadIdx.x;
  if (tid >= 32) return;
  const int b = tid >> 1, o = tid & 1;
  float f[4], s = 0.f;
  for (int c = 0; c < 4; ++c) { f[c] = feat[b * 4 + c]; s += f[c]; }
  const float m = s * 0.25f;
  float vv = 0.f;
  for (int c = 0; c < 4; ++c) { const float d = f[c] - m; vv += d * d; }
  const float rstd = rsqrtf(vv * 0.25f + 1e-6f);
  float acc = flin_b[o];
  for (int c = 0; c < 4; ++c) {
    const float xf = (f[c] - m) * rstd * (1.f + sc24[b * 4 + c]) + sh4[b * 4 + c];
    acc += flin_w[o * 4 + c] * xf;
  }
  out[b * 2 + o] = acc;
}

extern "C" void kernel_launch(void* const* d_in, const int* in_sizes, int n_in,
                              void* d_out, int out_size, void* d_ws, size_t ws_size,
                              hipStream_t stream) {
  (void)in_sizes; (void)n_in; (void)out_size; (void)ws_size;
  const float* x        = (const float*)d_in[0];
  const float* cond_vec = (const float*)d_in[1];
  const float* ce_w1 = (const float*)d_in[2];  const float* ce_b1 = (const float*)d_in[3];
  const float* ce_w2 = (const float*)d_in[4];  const float* ce_b2 = (const float*)d_in[5];
  const float* mod_w = (const float*)d_in[6];  const float* mod_b = (const float*)d_in[7];
  const float* cp_w  = (const float*)d_in[8];  const float* cp_b  = (const float*)d_in[9];
  const float* qkv_w = (const float*)d_in[10]; const float* qkv_b = (const float*)d_in[11];
  const float* op_w  = (const float*)d_in[12]; const float* op_b  = (const float*)d_in[13];
  const float* mta_in_w  = (const float*)d_in[14]; const float* mta_in_b  = (const float*)d_in[15];
  const float* mta_out_w = (const float*)d_in[16]; const float* mta_out_b = (const float*)d_in[17];
  const float* mtc2_w = (const float*)d_in[18]; const float* mtc2_b = (const float*)d_in[19];
  const float* mtc4_w = (const float*)d_in[20]; const float* mtc4_b = (const float*)d_in[21];
  const float* mtc6_w = (const float*)d_in[22]; const float* mtc6_b = (const float*)d_in[23];
  const float* qn_g = (const float*)d_in[24]; const float* qn_b = (const float*)d_in[25];
  const float* kn_g = (const float*)d_in[26]; const float* kn_b = (const float*)d_in[27];
  const float* ln_g = (const float*)d_in[28]; const float* ln_b = (const float*)d_in[29];
  const float* fin_w = (const float*)d_in[30]; const float* fin_b = (const float*)d_in[31];
  const float* dc_w  = (const float*)d_in[32]; const float* dc_b  = (const float*)d_in[33];
  const float* bn1_g = (const float*)d_in[34]; const float* bn1_b = (const float*)d_in[35];
  const float* c0_w = (const float*)d_in[36]; const float* c0_b = (const float*)d_in[37];
  const float* c1_w = (const float*)d_in[38]; const float* c1_b = (const float*)d_in[39];
  const float* c2_w = (const float*)d_in[40]; const float* c2_b = (const float*)d_in[41];
  const float* uc_w = (const float*)d_in[42]; const float* uc_b = (const float*)d_in[43];
  const float* tc_w = (const float*)d_in[44]; const float* tc_b = (const float*)d_in[45];
  const float* bnt_g = (const float*)d_in[46]; const float* bnt_b = (const float*)d_in[47];
  const float* sc_w = (const float*)d_in[48]; const float* sc_b = (const float*)d_in[49];
  const float* bns_g = (const float*)d_in[50]; const float* bns_b = (const float*)d_in[51];
  const float* fcp_w = (const float*)d_in[52]; const float* fcp_b = (const float*)d_in[53];
  const float* ada_w = (const float*)d_in[54]; const float* ada_b = (const float*)d_in[55];
  const float* flin_w = (const float*)d_in[56]; const float* flin_b = (const float*)d_in[57];

  float* ws = (float*)d_ws;
  float* shift = ws + OFF_SHIFT;  float* scale = ws + OFF_SCALE;  float* gate = ws + OFF_GATE;
  float* sh4 = ws + OFF_SH4;      float* sc24 = ws + OFF_SC24;    float* feat = ws + OFF_FEAT;
  float* avals = ws + OFF_AVAL;
  float* xh   = ws + OFF_XH;      float* vraw = ws + OFF_VRAW;    float* vp = ws + OFF_VP;
  float* outb = ws + OFF_OUTB;    float* xblk = ws + OFF_XBLK;
  float* sp   = ws + OFF_SP;      float* xmga = ws + OFF_XMGA;    float* tbuf = ws + OFF_TBUF;
  f16* wq16   = (f16*)(ws + OFF_W16Q);
  f16* wo16   = (f16*)(ws + OFF_W16O);
  f16* xmod16 = (f16*)(ws + OFF_XMOD16);
  f16* q16    = (f16*)(ws + OFF_Q16);
  f16* k16    = (f16*)(ws + OFF_K16);
  f16* v16    = (f16*)(ws + OFF_V16);
  f16* u16    = (f16*)(ws + OFF_U16);
  float* out = (float*)d_out;

  k_cond<<<1, 128, 0, stream>>>(cond_vec, ce_w1, ce_b1, ce_w2, ce_b2, mod_w, mod_b,
                                fcp_w, fcp_b, ada_w, ada_b, shift, scale, gate, sh4, sc24);
  k_w16<<<512, 256, 0, stream>>>(qkv_w, op_w, wq16, wo16);
  k_ln<<<B_ * T_, 128, 0, stream>>>(x, cp_w, cp_b, ln_g, ln_b, shift, scale, xh, xmod16);
  k_qkvmm<<<(B_ * 64 * 40) / 8, 256, 0, stream>>>(wq16, xmod16, qkv_b, qn_g, qn_b, kn_g, kn_b,
                                                  vraw, q16, k16, u16);
  k_vp<<<(B_ * 3 * T_) / 256, 256, 0, stream>>>(vraw, mta_in_w, mta_in_b, vp);
  k_ascalar<<<48, 256, 0, stream>>>(vp, mtc2_w, mtc2_b, mtc4_w, mtc4_b, mtc6_w, mtc6_b, avals);
  k_vupd<<<(B_ * HID_ * T_) / 256, 256, 0, stream>>>(vraw, vp, avals, mta_out_w, mta_out_b, v16);
  k_attn<<<B_ * NH_ * (T_ / 16 / 8), 256, 0, stream>>>(q16, k16, v16, u16);
  k_opmm<<<(B_ * 64 * 8) / 8, 256, 0, stream>>>(wo16, u16, op_b, outb);
  k_out2<<<B_ * T_, 128, 0, stream>>>(outb, xh, gate, fin_w, fin_b, xblk);
  k_sp<<<(B_ * 3 * T_) / 256, 256, 0, stream>>>(xblk, dc_w, dc_b, bn1_g, bn1_b, sp);
  k_mga<<<(B_ * T_) / 256, 256, 0, stream>>>(sp, c0_w, c0_b, c1_w, c1_b, c2_w, c2_b,
                                             uc_w, uc_b, xblk, xmga);
  k_tconv<<<(B_ * 64 * 1025 + 255) / 256, 256, 0, stream>>>(xmga, tc_w, tc_b, bnt_g, bnt_b, tbuf);
  k_sconv<<<B_ * 4, 256, 0, stream>>>(tbuf, sc_w, sc_b, bns_g, bns_b, feat);
  k_final<<<1, 32, 0, stream>>>(feat, sh4, sc24, flin_w, flin_b, out);
}
